// FourierBlock_1700807049903
// MI455X (gfx1250) — matrix operand order repaired and checked by reading, not compiled, not run on hardware
//
#include <hip/hip_runtime.h>
#include <hip/hip_bf16.h>
#include <math.h>

// ---------------------------------------------------------------------------
// FourierBlock for MI455X (gfx1250, wave32, WMMA 16x16x32 bf16, f32 accum).
// All heavy math is bf16 WMMA; spectral conv is expressed as 5 dense DFT
// GEMMs over the 32x64 retained modes (no FFT butterflies needed).
// Round-2 changes: templated + software-pipelined GEMM with 32x64 per-wave
// tiles; conv3x3 halo patch filled with global_load_async_to_lds_b128.
// ---------------------------------------------------------------------------

#define BSZ   8
#define CIN   64
#define COUT  64
#define HH    256
#define WW    256
#define MM1   32
#define MM2   32
#define HID   128
#define NPIX  (HH * WW)           // 65536

typedef __attribute__((ext_vector_type(16))) __bf16 v16bf;
typedef __attribute__((ext_vector_type(8)))  __bf16 v8bf;
typedef __attribute__((ext_vector_type(8)))  float  v8f;

__device__ __forceinline__ v16bf bf_concat(v8bf lo, v8bf hi) {
    return __builtin_shufflevector(lo, hi, 0,1,2,3,4,5,6,7,8,9,10,11,12,13,14,15);
}

// A fragment (16x32 bf16): lane<16 -> row=lane, K in {k0..k0+7, k0+16..k0+23};
// lane>=16 -> row=lane-16, K chunks shifted by 8.  (ISA 7.12.2)
__device__ __forceinline__ v16bf load_a_frag(const __bf16* tile, int lda, int k0, int lane) {
    int half = lane >> 4, row = lane & 15;
    const __bf16* p = tile + (long long)row * lda + k0 + half * 8;
    v8bf lo = *(const v8bf*)p;
    v8bf hi = *(const v8bf*)(p + 16);
    return bf_concat(lo, hi);
}

// Same fragment but converting from f32 storage on the fly (stage-1 reads x).
__device__ __forceinline__ v16bf load_a_frag_f32(const float* tile, int lda, int k0, int lane) {
    int half = lane >> 4, row = lane & 15;
    const float* p = tile + (long long)row * lda + k0 + half * 8;
    v16bf a;
#pragma unroll
    for (int i = 0; i < 8; ++i) { a[i] = (__bf16)p[i]; a[i + 8] = (__bf16)p[i + 16]; }
    return a;
}

// B fragment (32x16 bf16) from NT storage (Bt rows over K): lane<16 -> col n,
// K in {k0..k0+7, k0+16..k0+23}; lane>=16 -> K chunks shifted by 8.
__device__ __forceinline__ v16bf load_b_frag(const __bf16* rowp, int k0, int lane) {
    int half = lane >> 4;
    const __bf16* p = rowp + k0 + half * 8;
    v8bf lo = *(const v8bf*)p;
    v8bf hi = *(const v8bf*)(p + 16);
    return bf_concat(lo, hi);
}

__device__ __forceinline__ v8f wmma_bf16(v16bf a, v16bf b, v8f c) {
    return __builtin_amdgcn_wmma_f32_16x16x32_bf16(false, a, false, b, (short)0, c, false, false);
}

__device__ __forceinline__ float gelu_exact(float x) {
    return 0.5f * x * (1.0f + erff(x * 0.70710678118654752f));
}

// ---------------------------------------------------------------------------
// Generic batched NT-GEMM: C[b][M][N] (+)= A[b][M][K] * Bt[b][N][K]^T
// Per-wave tile: (16*MT) x 64; software-pipelined K loop (loads for k+1
// issued before WMMAs for k, so v_wmma overlaps global_load latency).
// M%(16*MT)==0, N%64==0, K%32==0 guaranteed by the driver.
// ---------------------------------------------------------------------------
template<int AF32, int ACC, int MT>
__global__ void __launch_bounds__(256) k_gemm_nt(
    const void* __restrict__ Av, int lda, long long sA,
    const __bf16* __restrict__ Bt, int ldb, long long sB,
    float* __restrict__ C, int ldc, long long sC,
    int M, int N, int K, int nbatch)
{
    int gw   = (int)((blockIdx.x * blockDim.x + threadIdx.x) >> 5);
    int lane = threadIdx.x & 31;
    int mtiles = M / (16 * MT), nstrips = N >> 6;
    int per_b  = mtiles * nstrips;
    long long total = (long long)nbatch * per_b;
    if (gw >= total) return;                       // wave-uniform: EXEC stays all-1s
    int bidx = gw / per_b;
    int r    = gw - bidx * per_b;
    int mt   = r / nstrips;
    int ns   = r - mt * nstrips;
    int half = lane >> 4, row = lane & 15;

    const __bf16* Ab = (const __bf16*)Av + bidx * sA + (long long)(mt * 16 * MT) * lda;
    const float*  Af = (const float*)Av  + bidx * sA + (long long)(mt * 16 * MT) * lda;
    const __bf16* Bb = Bt + bidx * sB + (long long)(ns * 64) * ldb;
    float*        Cb = C  + bidx * sC + (long long)(mt * 16 * MT) * ldc + ns * 64;

    v16bf a_cur[MT], b_cur[4];
#pragma unroll
    for (int i = 0; i < MT; ++i) {
        if constexpr (AF32) a_cur[i] = load_a_frag_f32(Af + (long long)i * 16 * lda, lda, 0, lane);
        else                a_cur[i] = load_a_frag   (Ab + (long long)i * 16 * lda, lda, 0, lane);
    }
#pragma unroll
    for (int t = 0; t < 4; ++t)
        b_cur[t] = load_b_frag(Bb + (long long)(t * 16 + row) * ldb, 0, lane);

    v8f acc[MT][4] = {};
    for (int k0 = 32; k0 < K; k0 += 32) {
        v16bf a_nxt[MT], b_nxt[4];
#pragma unroll
        for (int i = 0; i < MT; ++i) {
            if constexpr (AF32) a_nxt[i] = load_a_frag_f32(Af + (long long)i * 16 * lda, lda, k0, lane);
            else                a_nxt[i] = load_a_frag   (Ab + (long long)i * 16 * lda, lda, k0, lane);
        }
#pragma unroll
        for (int t = 0; t < 4; ++t)
            b_nxt[t] = load_b_frag(Bb + (long long)(t * 16 + row) * ldb, k0, lane);
        if (k0 + 32 < K) {
            if constexpr (AF32) __builtin_prefetch(Af + (long long)row * lda + k0 + 32, 0, 0);
            else                __builtin_prefetch(Ab + (long long)row * lda + k0 + 32, 0, 0);
        }
#pragma unroll
        for (int t = 0; t < 4; ++t)
#pragma unroll
            for (int i = 0; i < MT; ++i)
                acc[i][t] = wmma_bf16(a_cur[i], b_cur[t], acc[i][t]);
#pragma unroll
        for (int i = 0; i < MT; ++i) a_cur[i] = a_nxt[i];
#pragma unroll
        for (int t = 0; t < 4; ++t) b_cur[t] = b_nxt[t];
    }
#pragma unroll
    for (int t = 0; t < 4; ++t)
#pragma unroll
        for (int i = 0; i < MT; ++i)
            acc[i][t] = wmma_bf16(a_cur[i], b_cur[t], acc[i][t]);

    // C/D layout: lane -> N (lane&15), VGPR v -> M (= v + 8*(lane>=16)).
#pragma unroll
    for (int i = 0; i < MT; ++i)
#pragma unroll
        for (int t = 0; t < 4; ++t)
#pragma unroll
            for (int v = 0; v < 8; ++v) {
                long long off = (long long)(i * 16 + v + half * 8) * ldc + t * 16 + row;
                if constexpr (ACC) Cb[off] += acc[i][t][v];
                else               Cb[off]  = acc[i][t][v];
            }
}

// ---------------------------------------------------------------------------
// Fused pointwise MLP: out[b,o,hw] = W2 @ gelu(W1 @ x + b1) + b2
// Block: 128 pixels, 8 waves. Hidden tile transposed through LDS (bf16).
// xp is pixel-major bf16 x: [b][hw][c] (c contiguous -> NT rows for stage 1).
// ---------------------------------------------------------------------------
__global__ void __launch_bounds__(256) k_mlp(
    const __bf16* __restrict__ xp, const __bf16* __restrict__ w1b, const float* __restrict__ b1,
    const __bf16* __restrict__ w2b, const float* __restrict__ b2, float* __restrict__ out)
{
    __shared__ __align__(16) __bf16 hT[128 * 136];   // [pixel][hid], padded stride
    int tid = threadIdx.x, wave = tid >> 5, lane = tid & 31;
    int half = lane >> 4, row = lane & 15;
    long long pixBase = (long long)blockIdx.x * 128;

    // ---- stage 1: h[hid=128, n=128] = W1[128,64] * xp_tile[128,64]^T
    v8f acc[8] = {};
    for (int k0 = 0; k0 < CIN; k0 += 32) {
        v16bf a = load_a_frag(w1b + wave * 16 * CIN, CIN, k0, lane);
#pragma unroll
        for (int t = 0; t < 8; ++t) {
            const __bf16* bp = xp + (pixBase + t * 16 + row) * CIN;
            acc[t] = wmma_bf16(a, load_b_frag(bp, k0, lane), acc[t]);
        }
    }
#pragma unroll
    for (int t = 0; t < 8; ++t) {
#pragma unroll
        for (int v = 0; v < 8; ++v) {
            int m = wave * 16 + v + half * 8;          // hidden index
            int n = t * 16 + row;                      // pixel index
            float val = acc[t][v] + b1[m];
            hT[n * 136 + m] = (__bf16)gelu_exact(val);
        }
    }
    __syncthreads();

    // ---- stage 2: out[o=64, n=128] = W2[64,128] * hT[128,128]^T
    int mt2 = wave & 3, nh = wave >> 2;
    v8f acc2[4] = {};
    for (int k0 = 0; k0 < HID; k0 += 32) {
        v16bf a = load_a_frag(w2b + mt2 * 16 * HID, HID, k0, lane);
#pragma unroll
        for (int t = 0; t < 4; ++t) {
            const __bf16* bp = &hT[(nh * 64 + t * 16 + row) * 136];
            acc2[t] = wmma_bf16(a, load_b_frag(bp, k0, lane), acc2[t]);
        }
    }
#pragma unroll
    for (int t = 0; t < 4; ++t) {
#pragma unroll
        for (int v = 0; v < 8; ++v) {
            int o = mt2 * 16 + v + half * 8;
            int n = nh * 64 + t * 16 + row;
            long long P = pixBase + n;
            int b  = (int)(P >> 16);
            int hw = (int)(P & 65535);
            out[((long long)(b * COUT + o) << 16) + hw] = acc2[t][v] + b2[o];  // first writer
        }
    }
}

// ---------------------------------------------------------------------------
// conv3x3 SAME as implicit GEMM. Spatial tile 32x8 per block, halo patch in
// LDS (pixel-major bf16, zero padded). In-bounds pixels are pulled with the
// gfx1250 async copy (global_load_async_to_lds_b128, ASYNCcnt-tracked);
// OOB pixels get plain LDS zero stores. 9 taps x 2 c-chunks of WMMA.
// wc layout: [tap][o][c] bf16 (c contiguous -> A rows).
// ---------------------------------------------------------------------------
__global__ void __launch_bounds__(256) k_conv3x3(
    const __bf16* __restrict__ xp, const __bf16* __restrict__ wc,
    const float* __restrict__ bconv, float* __restrict__ out)
{
    __shared__ __align__(16) __bf16 patch[10 * 34 * 64];
    int tid = threadIdx.x;
    int tx = blockIdx.x, ty = blockIdx.y, b = blockIdx.z;

    for (int p = tid; p < 10 * 34; p += 256) {
        int py = p / 34, px = p - py * 34;
        int gy = ty * 8 + py - 1, gx = tx * 32 + px - 1;
        if (gy >= 0 && gy < HH && gx >= 0 && gx < WW) {
            // async copy 128 B (one pixel's 64 channels) straight into LDS
            unsigned int loff =
                (unsigned int)(uintptr_t)&patch[p * 64];   // low 32 bits of generic = LDS offset
            unsigned long long ga =
                (unsigned long long)(uintptr_t)(xp + (((long long)b << 16) + gy * WW + gx) * CIN);
#pragma unroll
            for (int i = 0; i < 8; ++i)
                asm volatile("global_load_async_to_lds_b128 %0, %1, off"
                             :: "v"(loff + (unsigned)(i * 16)), "v"(ga + (unsigned long long)(i * 16))
                             : "memory");
        } else {
            uint4* dst = (uint4*)&patch[p * 64];
            uint4 z = {0u, 0u, 0u, 0u};
#pragma unroll
            for (int i = 0; i < 8; ++i) dst[i] = z;
        }
    }
    asm volatile("s_wait_asynccnt 0x0" ::: "memory");
    __syncthreads();

    int wave = tid >> 5, lane = tid & 31, half = lane >> 4, row = lane & 15;
    int otile = wave & 3, nbase = (wave >> 2) * 128;

    v8f acc[8] = {};
    for (int tap = 0; tap < 9; ++tap) {
        int ky = tap / 3, kx = tap - ky * 3;
#pragma unroll
        for (int kc = 0; kc < 2; ++kc) {
            v16bf a = load_a_frag(wc + (tap * 64 + otile * 16) * 64, 64, kc * 32, lane);
#pragma unroll
            for (int t = 0; t < 8; ++t) {
                int n = nbase + t * 16 + row;
                int py = n >> 5, px = n & 31;
                const __bf16* bp = &patch[((py + ky) * 34 + (px + kx)) * 64];
                acc[t] = wmma_bf16(a, load_b_frag(bp, kc * 32, lane), acc[t]);
            }
        }
    }
#pragma unroll
    for (int t = 0; t < 8; ++t) {
#pragma unroll
        for (int v = 0; v < 8; ++v) {
            int o = otile * 16 + v + half * 8;
            int n = nbase + t * 16 + row;
            int py = n >> 5, px = n & 31;
            int gy = ty * 8 + py, gx = tx * 32 + px;
            long long idx = ((long long)(b * COUT + o) << 16) + gy * WW + gx;
            out[idx] += acc[t][v] + bconv[o];     // unique element per lane; serialized after MLP
        }
    }
}

// ---------------------------------------------------------------------------
// Prep: pixel-major bf16 copy of x  (xp[b][hw][c])
// ---------------------------------------------------------------------------
__global__ void __launch_bounds__(256) k_convert_xp(const float* __restrict__ x, __bf16* __restrict__ xp)
{
    int p = blockIdx.x * 256 + threadIdx.x;        // 0 .. B*NPIX-1
    int b = p >> 16, hw = p & 65535;
    __bf16 tmp[CIN];
#pragma unroll
    for (int c = 0; c < CIN; ++c)
        tmp[c] = (__bf16)x[(((long long)(b * CIN + c)) << 16) + hw];
    v8bf* dst = (v8bf*)(xp + (long long)p * CIN);
#pragma unroll
    for (int i = 0; i < 8; ++i) dst[i] = *(const v8bf*)(tmp + i * 8);
}

// ---------------------------------------------------------------------------
// Prep: small bf16 tables. Phases use exact integer mod-256 so no large-angle
// trig error.  Fw: forward W-DFT [64,256]; G2: forward H-DFT realified
// [128,512]; G4: inverse-H [512,128] (x 1/256); G5: inverse-rfft-W [256,64]
// (x 1/256, 2x for k>0); plus bf16 weight copies.
// ---------------------------------------------------------------------------
__global__ void __launch_bounds__(256) k_prep_small(
    const float* __restrict__ w_mlp1, const float* __restrict__ w_mlp2,
    const float* __restrict__ w_conv,
    __bf16* Fw, __bf16* G2, __bf16* G4, __bf16* G5,
    __bf16* w1b, __bf16* w2b, __bf16* wcb)
{
    int i = blockIdx.x * 256 + threadIdx.x;        // 0 .. 65535
    const float W0 = 6.28318530717958647692f / 256.0f;

    if (i < 64 * 256) {                            // Fw[m][w]
        int m = i >> 8, w = i & 255;
        int mm = (m < 32) ? m : m - 32;
        float ang = W0 * (float)((mm * w) & 255);
        Fw[i] = (__bf16)((m < 32) ? cosf(ang) : -sinf(ang));
    }
    if (i < 128 * 512) {                           // G2[n][k]
        int n = i >> 9, k = i & 511;
        int u = n & 63, isIm = n >> 6;
        int h = k & 255, kIm = k >> 8;
        int m1 = (u < 32) ? u : (192 + u);
        float ang = W0 * (float)((m1 * h) & 255);
        float cg = cosf(ang), sg = sinf(ang);
        // Yr row = [cos | sin], Yi row = [-sin | cos]   (E = cos - i sin)
        float val = isIm ? (kIm ? cg : -sg) : (kIm ? sg : cg);
        G2[i] = (__bf16)val;
    }
    if (i < 512 * 128) {                           // G4[n][k]
        int n = i >> 7, k = i & 127;
        int h = n & 255, isIm = n >> 8;
        int u = k & 63, kIm = k >> 6;
        int m1 = (u < 32) ? u : (192 + u);
        float ang = W0 * (float)((m1 * h) & 255);
        float cg = cosf(ang) * (1.0f / 256.0f), sg = sinf(ang) * (1.0f / 256.0f);
        // Zr row = [cos | -sin], Zi row = [sin | cos]   (E' = (cos + i sin)/256)
        float val = isIm ? (kIm ? cg : sg) : (kIm ? -sg : cg);
        G4[i] = (__bf16)val;
    }
    if (i < 256 * 64) {                            // G5[w][k]
        int w = i >> 6, k = i & 63;
        int m2 = k & 31, kIm = k >> 5;
        float ang = W0 * (float)((m2 * w) & 255);
        float sc = (m2 == 0) ? (1.0f / 256.0f) : (2.0f / 256.0f);
        float val = kIm ? ((m2 == 0) ? 0.0f : -sc * sinf(ang)) : sc * cosf(ang);
        G5[i] = (__bf16)val;
    }
    if (i < 9 * 64 * 64) {                         // wcb[tap][o][c] <- w_conv[o][c][ky][kx]
        int tap = i >> 12, o = (i >> 6) & 63, c = i & 63;
        wcb[i] = (__bf16)w_conv[(o * 64 + c) * 9 + tap];
    }
    if (i < HID * CIN)  w1b[i] = (__bf16)w_mlp1[i];
    if (i < COUT * HID) w2b[i] = (__bf16)w_mlp2[i];
}

// Per-mode realified complex weights: Gw[mode=(u,m2)][n=128][k=128]
// rows n=o:    [ Wr[c,o] | -Wi[c,o] ]  -> Or
// rows n=64+o: [ Wi[c,o] |  Wr[c,o] ]  -> Oi
__global__ void __launch_bounds__(256) k_prep_gw(
    const float* __restrict__ w1r, const float* __restrict__ w1i,
    const float* __restrict__ w2r, const float* __restrict__ w2i,
    __bf16* __restrict__ Gw)
{
    long long i = (long long)blockIdx.x * 256 + threadIdx.x;   // < 2048*16384
    int mode = (int)(i >> 14);
    int n = ((int)(i >> 7)) & 127;
    int k = (int)i & 127;
    int u = mode >> 5, m2 = mode & 31;
    int o = n & 63, nIm = n >> 6;
    int c = k & 63, kIm = k >> 6;
    const float* Wr = (u < 32) ? w1r : w2r;
    const float* Wi = (u < 32) ? w1i : w2i;
    int mu = (u < 32) ? u : (u - 32);
    int widx = ((c * COUT + o) * MM1 + mu) * MM2 + m2;
    float vr = Wr[widx], vi = Wi[widx];
    float val = nIm ? (kIm ? vr : vi) : (kIm ? -vi : vr);
    Gw[i] = (__bf16)val;
}

// ---------------------------------------------------------------------------
// Transposes between spectral stages (f32 -> bf16 layout shuffles).
// ---------------------------------------------------------------------------
// T1: C1[(b,c,h)][64] -> X2[(b,c,m)][512] = [Re(h) | Im(h)]
__global__ void __launch_bounds__(256) k_t1(const float* __restrict__ C1, __bf16* __restrict__ X2)
{
    int idx = blockIdx.x * 256 + threadIdx.x;      // b*c*m*h = 4194304
    int h = idx & 255, m = (idx >> 8) & 31, bc = idx >> 13;
    long long src = ((long long)(bc << 8) + h) * 64 + m;
    long long dst = ((long long)(bc * 32 + m) << 9) + h;
    X2[dst]       = (__bf16)C1[src];
    X2[dst + 256] = (__bf16)C1[src + 32];
}
// T2: C2[(b,c,m2)][128] -> A3[mode=(u,m2)][16][128]; rows 8..15 zero.
__global__ void __launch_bounds__(256) k_t2(const float* __restrict__ C2, __bf16* __restrict__ A3)
{
    int idx = blockIdx.x * 256 + threadIdx.x;      // 2048*16*128 = 4194304
    int k = idx & 127, rb = (idx >> 7) & 15, mode = idx >> 11;
    int u = mode >> 5, m2 = mode & 31;
    float val = 0.0f;
    if (rb < 8) {
        int c = k & 63, kIm = k >> 6;
        val = C2[((long long)((rb * CIN + c) * MM2 + m2)) * 128 + (kIm ? 64 + u : u)];
    }
    A3[(long long)idx] = (__bf16)val;
}
// T3: C3[mode][16][128] -> A4[(b,o,m2)][128] = [Or(u) | Oi(u)]
__global__ void __launch_bounds__(256) k_t3(const float* __restrict__ C3, __bf16* __restrict__ A4)
{
    int idx = blockIdx.x * 256 + threadIdx.x;      // 8*64*32*128 = 2097152
    int k = idx & 127, m2 = (idx >> 7) & 31, o = (idx >> 12) & 63, b = idx >> 18;
    int u = k & 63, kIm = k >> 6;
    int mode = u * 32 + m2;
    float val = C3[(long long)mode * 2048 + b * 128 + (kIm ? 64 + o : o)];
    A4[(long long)idx] = (__bf16)val;
}
// T4: C4[(b,o,m2)][512] -> A5[(b,o,h)][64] = [Zr(m2) | Zi(m2)]
__global__ void __launch_bounds__(256) k_t4(const float* __restrict__ C4, __bf16* __restrict__ A5)
{
    int idx = blockIdx.x * 256 + threadIdx.x;      // 8*64*256*32 = 4194304
    int m2 = idx & 31, h = (idx >> 5) & 255, o = (idx >> 13) & 63, b = idx >> 19;
    long long src = ((long long)((b * COUT + o) * MM2 + m2)) * 512 + h;
    long long dst = ((long long)((b * COUT + o) * HH + h)) * 64 + m2;
    A5[dst]      = (__bf16)C4[src];
    A5[dst + 32] = (__bf16)C4[src + 256];
}

// ---------------------------------------------------------------------------
extern "C" void kernel_launch(void* const* d_in, const int* in_sizes, int n_in,
                              void* d_out, int out_size, void* d_ws, size_t ws_size,
                              hipStream_t stream)
{
    (void)in_sizes; (void)n_in; (void)out_size; (void)ws_size;

    const float* x      = (const float*)d_in[0];
    const float* w1r    = (const float*)d_in[1];
    const float* w1i    = (const float*)d_in[2];
    const float* w2r    = (const float*)d_in[3];
    const float* w2i    = (const float*)d_in[4];
    const float* w_mlp1 = (const float*)d_in[5];
    const float* b_mlp1 = (const float*)d_in[6];
    const float* w_mlp2 = (const float*)d_in[7];
    const float* b_mlp2 = (const float*)d_in[8];
    const float* w_conv = (const float*)d_in[9];
    const float* b_conv = (const float*)d_in[10];
    float* out = (float*)d_out;

    // ---- workspace layout (bump allocator, 256 B aligned). ~190 MB total.
    char* W = (char*)d_ws;
    size_t off = 0;
    auto alloc = [&](size_t bytes) -> char* {
        char* p = W + off;
        off += (bytes + 255) & ~(size_t)255;
        return p;
    };
    __bf16* xp  = (__bf16*)alloc((size_t)BSZ * NPIX * CIN * 2);        // 64 MB pixel-major x
    __bf16* Gw  = (__bf16*)alloc((size_t)2048 * 128 * 128 * 2);        // 67 MB per-mode weights
    float*  P   = (float*) alloc((size_t)33554432);                    // ping (C1/C2/C3/C4)
    char*   Q   =          alloc((size_t)16777216);                    // pong (X2/A3/A4/A5)
    __bf16* Fw  = (__bf16*)alloc(64  * 256 * 2);
    __bf16* G2  = (__bf16*)alloc(128 * 512 * 2);
    __bf16* G4  = (__bf16*)alloc(512 * 128 * 2);
    __bf16* G5  = (__bf16*)alloc(256 * 64  * 2);
    __bf16* wcb = (__bf16*)alloc(9 * 64 * 64 * 2);
    __bf16* w1b = (__bf16*)alloc(HID * CIN * 2);
    __bf16* w2b = (__bf16*)alloc(COUT * HID * 2);

    // ---- prep
    k_convert_xp<<<(BSZ * NPIX) / 256, 256, 0, stream>>>(x, xp);
    k_prep_small<<<256, 256, 0, stream>>>(w_mlp1, w_mlp2, w_conv, Fw, G2, G4, G5, w1b, w2b, wcb);
    k_prep_gw<<<(2048 * 16384) / 256, 256, 0, stream>>>(w1r, w1i, w2r, w2i, Gw);

    // ---- MLP branch (writes out with '='), then conv branch (accumulates)
    k_mlp<<<(BSZ * NPIX) / 128, 256, 0, stream>>>(xp, w1b, b_mlp1, w2b, b_mlp2, out);
    k_conv3x3<<<dim3(WW / 32, HH / 8, BSZ), 256, 0, stream>>>(xp, wcb, b_conv, out);

    // ---- spectral branch: 5 NT-GEMMs + 4 transposes (P/Q ping-pong)
    auto gemm = [&](const void* A, int lda, long long sA, int af32,
                    const __bf16* Bt, int ldb, long long sB,
                    float* Cp, int ldc, long long sC,
                    int M, int N, int K, int nb, int acc, int mt2) {
        int mrows = mt2 ? 32 : 16;
        long long waves = (long long)nb * (M / mrows) * (N >> 6);
        int blocks = (int)((waves + 7) / 8);
        if (af32)
            k_gemm_nt<1, 0, 2><<<blocks, 256, 0, stream>>>(A, lda, sA, Bt, ldb, sB, Cp, ldc, sC, M, N, K, nb);
        else if (acc)
            k_gemm_nt<0, 1, 2><<<blocks, 256, 0, stream>>>(A, lda, sA, Bt, ldb, sB, Cp, ldc, sC, M, N, K, nb);
        else if (mt2)
            k_gemm_nt<0, 0, 2><<<blocks, 256, 0, stream>>>(A, lda, sA, Bt, ldb, sB, Cp, ldc, sC, M, N, K, nb);
        else
            k_gemm_nt<0, 0, 1><<<blocks, 256, 0, stream>>>(A, lda, sA, Bt, ldb, sB, Cp, ldc, sC, M, N, K, nb);
    };

    // S1: Xw[(b,c,h)][64] = x[(b,c,h)][w=256] * Fw^T       (f32 A, convert on load)
    gemm(x, WW, 0, 1, Fw, WW, 0, P, 64, 0, BSZ * CIN * HH, 64, WW, 1, 0, 1);
    k_t1<<<4194304 / 256, 256, 0, stream>>>(P, (__bf16*)Q);
    // S2: Y[(b,c,m2)][128] = X2[(b,c,m2)][512] * G2^T
    gemm(Q, 512, 0, 0, G2, 512, 0, P, 128, 0, BSZ * CIN * MM2, 128, 512, 1, 0, 1);
    k_t2<<<4194304 / 256, 256, 0, stream>>>(P, (__bf16*)Q);
    // S3: per-mode complex mix: C3[mode][16][128] = A3[mode][16][128] * Gw[mode]^T
    gemm(Q, 128, 2048, 0, Gw, 128, 16384, P, 128, 2048, 16, 128, 128, 2048, 0, 0);
    k_t3<<<2097152 / 256, 256, 0, stream>>>(P, (__bf16*)Q);
    // S4: Z[(b,o,m2)][512] = A4[(b,o,m2)][128] * G4^T      (inverse-H)
    gemm(Q, 128, 0, 0, G4, 128, 0, P, 512, 0, BSZ * COUT * MM2, 512, 128, 1, 0, 1);
    k_t4<<<4194304 / 256, 256, 0, stream>>>(P, (__bf16*)Q);
    // S5: out[(b,o,h)][w=256] += A5[(b,o,h)][64] * G5^T    (inverse-rfft-W)
    gemm(Q, 64, 0, 0, G5, 64, 0, out, WW, 0, BSZ * COUT * HH, WW, 64, 1, 1, 1);
}
